// Net_31722628448620
// MI455X (gfx1250) — matrix operand order repaired and checked by reading, not compiled, not run on hardware
//
#include <hip/hip_runtime.h>

// ---------------------------------------------------------------------------
// 2-layer bidirectional LSTM + 2 linear heads for MI455X (gfx1250, wave32).
// - All matmuls: v_wmma_f32_16x16x32_f16 (f16 operands, fp32 accumulate).
// - Recurrent scan: Whh^T resident in VGPRs, h in LDS, and the precomputed
//   gate slab streamed one step ahead via global_load_async_to_lds_b128
//   (ASYNCcnt double-buffering) so global latency is off the critical path.
// - GEMM: one wave computes a 16x64 C strip (4x A-operand reuse).
// ---------------------------------------------------------------------------

typedef __attribute__((ext_vector_type(16))) _Float16 v16h;
typedef __attribute__((ext_vector_type(8)))  _Float16 v8h;
typedef __attribute__((ext_vector_type(8)))  float    v8f;

#define Tt   1024
#define Bb   64
#define Ee   32
#define Hh   100
#define G4H  400          // 4*H
#define KHP  128          // H padded to K multiple of 32
#define LD1  224          // 2H=200 padded row stride (layer outputs)
#define LDG  416          // 4H=400 padded row stride (g-head input)
#define MTB  (Tt*Bb)      // 65536 rows
#define MG   (Bb*(Tt-1))  // 65472 rows
#define FOUT 20
#define GOUT 400
#define NP4H 448          // 400 padded to N multiple of 64
#define NPF  64           // 20  padded to N multiple of 64

// --- ws layout (bytes, all 256-aligned) ---
#define OFF_ZF    ((size_t)0)
#define SZ_Z      ((size_t)MTB*G4H*2)                 // f16 gate buffer, 52,428,800
#define OFF_ZB    (OFF_ZF + SZ_Z)
#define OFF_X     (OFF_ZB + SZ_Z)                     // [T*B,32] f16
#define SZ_X      ((size_t)MTB*Ee*2)
#define OFF_OUT0  (OFF_X + SZ_X)                      // [T*B,224] f16 (time-major)
#define SZ_OUT    ((size_t)MTB*LD1*2)
#define OFF_OUT1  (OFF_OUT0 + SZ_OUT)                 // [B*T,224] f16 (batch-major)
#define OFF_GIN   (OFF_OUT1 + SZ_OUT)                 // [65472,416] f16
#define SZ_GIN    ((size_t)MG*LDG*2)
#define OFF_W     (OFF_GIN + SZ_GIN)
#define OFF_WIH0F (OFF_W)
#define OFF_WIH0B (OFF_WIH0F + (size_t)NP4H*Ee*2)
#define OFF_WHH0F (OFF_WIH0B + (size_t)NP4H*Ee*2)
#define OFF_WHH0B (OFF_WHH0F + (size_t)G4H*KHP*2)
#define OFF_WHH1F (OFF_WHH0B + (size_t)G4H*KHP*2)
#define OFF_WHH1B (OFF_WHH1F + (size_t)G4H*KHP*2)
#define OFF_WIH1F (OFF_WHH1B + (size_t)G4H*KHP*2)
#define OFF_WIH1B (OFF_WIH1F + (size_t)NP4H*LD1*2)
#define OFF_FW    (OFF_WIH1B + (size_t)NP4H*LD1*2)    // [64,224]
#define OFF_GW    (OFF_FW + (size_t)NPF*LD1*2)        // [448,416]

// Scan LDS: zS fp32 [64][400] + hA f16 [64][128] + Zbuf f16 [2][64][400]
#define ZS_BYTES   (Bb*G4H*4)     // 102,400
#define HA_BYTES   (Bb*KHP*2)     //  16,384
#define ZSLAB      (Bb*G4H*2)     //  51,200 per time step (f16)
#define SCAN_SMEM  (ZS_BYTES + HA_BYTES + 2*ZSLAB)    // 221,184 B (<320KB WGP LDS)

// Assemble a v16h A/B operand from two contiguous 16B half8 chunks.
__device__ __forceinline__ v16h ld16(const _Float16* p0, const _Float16* p1) {
  v8h lo = *(const v8h*)p0;
  v8h hi = *(const v8h*)p1;
  v16h r;
#pragma unroll
  for (int i = 0; i < 8; ++i) { r[i] = lo[i]; r[8 + i] = hi[i]; }
  return r;
}

// CDNA5 async global->LDS copy (16B per lane), tracked by ASYNCcnt.
__device__ __forceinline__ void async_copy16(void* lds_dst, const void* gsrc) {
  __attribute__((address_space(3))) char* l =
      (__attribute__((address_space(3))) char*)lds_dst;
  asm volatile("global_load_async_to_lds_b128 %0, %1, off"
               :: "v"(l), "v"(gsrc) : "memory");
}
__device__ __forceinline__ void wait_async0() {
  asm volatile("s_wait_asynccnt 0x0" ::: "memory");
}

__device__ __forceinline__ float sigm(float x) {
  return 1.0f / (1.0f + __expf(-x));
}
__device__ __forceinline__ float tanh_fast(float x) {
  float a = fabsf(x);
  float e = __expf(-2.0f * a);
  float t = (1.0f - e) / (1.0f + e);
  return copysignf(t, x);
}

// ---------------- small prep kernels ----------------

__global__ void pad_to_f16(const float* __restrict__ s, int sR, int sC,
                           _Float16* __restrict__ d, int dR, int dC) {
  int i = blockIdx.x * blockDim.x + threadIdx.x;
  if (i >= dR * dC) return;
  int r = i / dC, c = i - r * dC;
  float v = (r < sR && c < sC) ? s[r * sC + c] : 0.0f;
  d[i] = (_Float16)v;
}

__global__ void fill_zero_f16(_Float16* __restrict__ p, int n) {
  int i = blockIdx.x * blockDim.x + threadIdx.x;
  if (i < n) p[i] = (_Float16)0.0f;
}

// X[t*B+b, e] = emb[tokens[b,t], e]  (f16)
__global__ void embed_k(const int* __restrict__ tokens, const float* __restrict__ emb,
                        _Float16* __restrict__ X) {
  int i = blockIdx.x * blockDim.x + threadIdx.x;   // T*B*E = 2M
  if (i >= MTB * Ee) return;
  int e  = i & (Ee - 1);
  int tb = i >> 5;              // t*64 + b
  int b  = tb & (Bb - 1);
  int t  = tb >> 6;
  int tok = tokens[b * Tt + t];
  X[i] = (_Float16)emb[tok * Ee + e];
}

// Gin[b*(T-1)+t, :] = out1[b*T+t, 0:200] ++ out1[b*T+t+1, 0:200] ++ zeros(16)
__global__ void build_gin(const _Float16* __restrict__ out1, _Float16* __restrict__ G) {
  int i = blockIdx.x * blockDim.x + threadIdx.x;   // 65472*416
  if (i >= MG * LDG) return;
  int c = i % LDG;
  int r = i / LDG;
  int b = r / (Tt - 1), t = r - b * (Tt - 1);
  _Float16 v = (_Float16)0.0f;
  if (c < 200)      v = out1[((size_t)(b * Tt + t))     * LD1 + c];
  else if (c < 400) v = out1[((size_t)(b * Tt + t + 1)) * LD1 + (c - 200)];
  G[i] = v;
}

// ---------------- generic WMMA GEMM: C[M,N] = A[M,K] * Bw[N,K]^T + bias ----------------
// One wave computes a 16x64 C strip (4 N-tiles): A loaded once per K-chunk and
// reused across 4 WMMAs. K = Kc*32, M multiple of 16, Bw N-padded to 64, N store-masked.
__global__ void __launch_bounds__(32)
gemm_wmma(const _Float16* __restrict__ A, int lda,
          const _Float16* __restrict__ Bw, int ldb,
          const float* __restrict__ bias,
          void* __restrict__ Cout, int ldc,
          int Kc, int Nvalid, int storeF16) {
  const int lane = threadIdx.x & 31;
  const int l = lane & 15, g = lane >> 4;
  const size_t m0 = (size_t)blockIdx.x * 16;
  const int n0 = blockIdx.y * 64;

  v8f acc[4] = {};
  const _Float16* aRow = A + (m0 + (size_t)l) * lda;
  const _Float16* bRow = Bw + ((size_t)(n0 + l)) * ldb + g * 8;
  for (int kc = 0; kc < Kc; ++kc) {
    const _Float16* ap = aRow + kc * 32 + g * 8;
    v16h a = ld16(ap, ap + 16);       // K 0-7/16-23 (g=0) or 8-15/24-31 (g=1)
#pragma unroll
    for (int q = 0; q < 4; ++q) {
      const _Float16* bp = bRow + (size_t)(q * 16) * ldb + kc * 32;
      v16h b = ld16(bp, bp + 16);     // K 0-7/16-23 (g=0) or 8-15/24-31 (g=1) of weight row n
      acc[q] = __builtin_amdgcn_wmma_f32_16x16x32_f16(false, a, false, b,
                                                      (short)0, acc[q], false, false);
    }
  }
#pragma unroll
  for (int q = 0; q < 4; ++q) {
    int n = n0 + q * 16 + l;
    if (n < Nvalid) {
      float bv = bias ? bias[n] : 0.0f;
#pragma unroll
      for (int j = 0; j < 8; ++j) {
        size_t m = m0 + j + g * 8;    // C layout: VGPR j -> M = j (+8 upper half)
        float v = acc[q][j] + bv;
        if (storeF16) ((_Float16*)Cout)[m * (size_t)ldc + n] = (_Float16)v;
        else          ((float*)Cout)[m * (size_t)ldc + n] = v;
      }
    }
  }
}

// ---------------- recurrent LSTM scan (one block per direction) ----------------
// Per step: z[64,400] = Zpre[t] (async-staged in LDS) + h @ Whh^T (WMMA),
// then fused gate update. Wave w of 25 owns gate columns [16w,16w+16); its
// Whh^T operands stay in VGPRs for all 1024 steps.
__global__ void __launch_bounds__(800)
lstm_scan(const _Float16* __restrict__ Zf, const _Float16* __restrict__ Zb,
          const _Float16* __restrict__ WhhF, const _Float16* __restrict__ WhhB,
          const float* __restrict__ h0L, const float* __restrict__ c0L,
          _Float16* __restrict__ outH, int timeMajorOut) {
  extern __shared__ char smem[];
  float*    zS   = (float*)smem;                        // [64][400] gate pre-acts
  _Float16* hA   = (_Float16*)(smem + ZS_BYTES);        // [64][128] h (f16, K-pad)
  char*     zBuf = smem + ZS_BYTES + HA_BYTES;          // [2][51,200B] Z double buffer

  const int dir = blockIdx.x;
  const _Float16* Zg  = dir ? Zb   : Zf;
  const _Float16* Whh = dir ? WhhB : WhhF;
  const float*    h0  = h0L + dir * Bb * Hh;
  const float*    c0  = c0L + dir * Bb * Hh;

  const int tid  = threadIdx.x;
  const int lane = tid & 31, w = tid >> 5;              // w in [0,25)
  const int l = lane & 15, g = lane >> 4;

  // Preload B operands: Whh^T tiles for this wave's 16 output columns, 4 K-chunks.
  v16h Bt[4];
  {
    const _Float16* br = Whh + (size_t)(w * 16 + l) * KHP + g * 8;
#pragma unroll
    for (int kc = 0; kc < 4; ++kc)
      Bt[kc] = ld16(br + kc * 32, br + kc * 32 + 16);
  }

  // Init hidden/cell state. 800 threads x 8 cells = 6400 = 64x100.
  for (int i = tid; i < Bb * KHP; i += 800) hA[i] = (_Float16)0.0f;
  __syncthreads();
  int bArr[8], jArr[8];
  float cr[8];
#pragma unroll
  for (int u = 0; u < 8; ++u) {
    int cell = tid * 8 + u;
    int b = cell / Hh, j = cell - b * Hh;
    bArr[u] = b; jArr[u] = j;
    cr[u] = c0[b * Hh + j];
    hA[b * KHP + j] = (_Float16)h0[b * Hh + j];
  }

  // Stage first Z slab into buffer 0 (3200 x 16B chunks, 4 per thread).
  {
    const int t0 = dir ? (Tt - 1) : 0;
    const char* src = (const char*)(Zg + (size_t)t0 * Bb * G4H);
#pragma unroll
    for (int k = 0; k < 4; ++k) {
      int off = (tid + k * 800) * 16;
      async_copy16(zBuf + off, src + off);
    }
  }
  wait_async0();
  __syncthreads();

  int cur = 0;
  for (int step = 0; step < Tt; ++step) {
    const int t = dir ? (Tt - 1 - step) : step;

    // Kick off async staging of the next step's slab into the spare buffer.
    if (step < Tt - 1) {
      const int tn = dir ? (t - 1) : (t + 1);
      const char* src = (const char*)(Zg + (size_t)tn * Bb * G4H);
      char* dst = zBuf + (cur ^ 1) * ZSLAB;
#pragma unroll
      for (int k = 0; k < 4; ++k) {
        int off = (tid + k * 800) * 16;
        async_copy16(dst + off, src + off);
      }
    }

    // z = Zpre + h @ Whh^T  (4 M-tiles x this wave's N-tile, K=128)
    const _Float16* zpre = (const _Float16*)(zBuf + cur * ZSLAB);
    const int n = w * 16 + l;
#pragma unroll
    for (int mt = 0; mt < 4; ++mt) {
      v8f acc = {};
#pragma unroll
      for (int kc = 0; kc < 4; ++kc) {
        const _Float16* ap = hA + (mt * 16 + l) * KHP + kc * 32 + g * 8;
        v16h a = ld16(ap, ap + 16);
        acc = __builtin_amdgcn_wmma_f32_16x16x32_f16(false, a, false, Bt[kc],
                                                     (short)0, acc, false, false);
      }
#pragma unroll
      for (int j = 0; j < 8; ++j) {
        int m = mt * 16 + j + g * 8;
        zS[m * G4H + n] = acc[j] + (float)zpre[m * G4H + n];
      }
    }
    __syncthreads();

    // Fused gate update: i,f,g,o at columns j, 100+j, 200+j, 300+j.
#pragma unroll
    for (int u = 0; u < 8; ++u) {
      int b = bArr[u], j = jArr[u];
      const float* zb = zS + b * G4H;
      float iv = zb[j], fv = zb[Hh + j], gv = zb[2 * Hh + j], ov = zb[3 * Hh + j];
      float cn = sigm(fv) * cr[u] + sigm(iv) * tanh_fast(gv);
      cr[u] = cn;
      float hv = sigm(ov) * tanh_fast(cn);
      hA[b * KHP + j] = (_Float16)hv;
      size_t row = timeMajorOut ? (size_t)(t * Bb + b) : (size_t)(b * Tt + t);
      outH[row * LD1 + dir * Hh + j] = (_Float16)hv;
    }
    wait_async0();     // next slab resident in LDS before anyone reads it
    __syncthreads();
    cur ^= 1;
  }
}

// ---------------- host orchestration ----------------

extern "C" void kernel_launch(void* const* d_in, const int* in_sizes, int n_in,
                              void* d_out, int out_size, void* d_ws, size_t ws_size,
                              hipStream_t stream) {
  const int*   tokens = (const int*)  d_in[0];
  const float* emb    = (const float*)d_in[2];
  const float* Wih0f  = (const float*)d_in[3];
  const float* Whh0f  = (const float*)d_in[4];
  const float* b0f    = (const float*)d_in[5];
  const float* Wih0b  = (const float*)d_in[6];
  const float* Whh0b  = (const float*)d_in[7];
  const float* b0b    = (const float*)d_in[8];
  const float* Wih1f  = (const float*)d_in[9];
  const float* Whh1f  = (const float*)d_in[10];
  const float* b1f    = (const float*)d_in[11];
  const float* Wih1b  = (const float*)d_in[12];
  const float* Whh1b  = (const float*)d_in[13];
  const float* b1b    = (const float*)d_in[14];
  const float* h0     = (const float*)d_in[15];   // [4,B,H]
  const float* c0     = (const float*)d_in[16];
  const float* f_w    = (const float*)d_in[17];
  const float* f_b    = (const float*)d_in[18];
  const float* g_w    = (const float*)d_in[19];
  const float* g_b    = (const float*)d_in[20];
  float* out = (float*)d_out;

  char* ws = (char*)d_ws;
  _Float16* Zf    = (_Float16*)(ws + OFF_ZF);
  _Float16* Zb    = (_Float16*)(ws + OFF_ZB);
  _Float16* X     = (_Float16*)(ws + OFF_X);
  _Float16* out0  = (_Float16*)(ws + OFF_OUT0);
  _Float16* out1  = (_Float16*)(ws + OFF_OUT1);
  _Float16* Gin   = (_Float16*)(ws + OFF_GIN);
  _Float16* wih0f = (_Float16*)(ws + OFF_WIH0F);
  _Float16* wih0b = (_Float16*)(ws + OFF_WIH0B);
  _Float16* whh0f = (_Float16*)(ws + OFF_WHH0F);
  _Float16* whh0b = (_Float16*)(ws + OFF_WHH0B);
  _Float16* whh1f = (_Float16*)(ws + OFF_WHH1F);
  _Float16* whh1b = (_Float16*)(ws + OFF_WHH1B);
  _Float16* wih1f = (_Float16*)(ws + OFF_WIH1F);
  _Float16* wih1b = (_Float16*)(ws + OFF_WIH1B);
  _Float16* fwh   = (_Float16*)(ws + OFF_FW);
  _Float16* gwh   = (_Float16*)(ws + OFF_GW);

  auto pad = [&](const float* s, int sR, int sC, _Float16* d, int dR, int dC) {
    int n = dR * dC;
    pad_to_f16<<<(n + 255) / 256, 256, 0, stream>>>(s, sR, sC, d, dR, dC);
  };

  // Weight conversion / padding (f16; K zero-padded to x32, N to x64 for GEMM B).
  pad(Wih0f, G4H, Ee,  wih0f, NP4H, Ee);
  pad(Wih0b, G4H, Ee,  wih0b, NP4H, Ee);
  pad(Whh0f, G4H, Hh,  whh0f, G4H,  KHP);
  pad(Whh0b, G4H, Hh,  whh0b, G4H,  KHP);
  pad(Whh1f, G4H, Hh,  whh1f, G4H,  KHP);
  pad(Whh1b, G4H, Hh,  whh1b, G4H,  KHP);
  pad(Wih1f, G4H, 200, wih1f, NP4H, LD1);
  pad(Wih1b, G4H, 200, wih1b, NP4H, LD1);
  pad(f_w,   FOUT, 200, fwh, NPF,  LD1);
  pad(g_w,   GOUT, 400, gwh, NP4H, LDG);

  // Zero out0+out1 (contiguous) so K-pad columns are zero.
  {
    int n = 2 * MTB * LD1;
    fill_zero_f16<<<(n + 255) / 256, 256, 0, stream>>>(out0, n);
  }

  // Embedding gather -> f16 [T*B, 32].
  embed_k<<<(MTB * Ee + 255) / 256, 256, 0, stream>>>(tokens, emb, X);

  // Layer 0 input projections: Z = X @ Wih^T + b, f16 [T*B, 400].
  gemm_wmma<<<dim3(MTB / 16, NP4H / 64), 32, 0, stream>>>(X, Ee, wih0f, Ee, b0f, Zf, G4H, 1, G4H, 1);
  gemm_wmma<<<dim3(MTB / 16, NP4H / 64), 32, 0, stream>>>(X, Ee, wih0b, Ee, b0b, Zb, G4H, 1, G4H, 1);

  // Layer 0 scan (time-major out0 for layer-1 GEMM).
  (void)hipFuncSetAttribute(reinterpret_cast<const void*>(lstm_scan),
                            hipFuncAttributeMaxDynamicSharedMemorySize, SCAN_SMEM);
  lstm_scan<<<2, 800, SCAN_SMEM, stream>>>(Zf, Zb, whh0f, whh0b,
                                           h0 + 0 * Bb * Hh, c0 + 0 * Bb * Hh, out0, 1);

  // Layer 1 input projections (reuse Z buffers): Z = out0 @ Wih1^T + b.
  gemm_wmma<<<dim3(MTB / 16, NP4H / 64), 32, 0, stream>>>(out0, LD1, wih1f, LD1, b1f, Zf, G4H, LD1 / 32, G4H, 1);
  gemm_wmma<<<dim3(MTB / 16, NP4H / 64), 32, 0, stream>>>(out0, LD1, wih1b, LD1, b1b, Zb, G4H, LD1 / 32, G4H, 1);

  // Layer 1 scan (batch-major out1 for the heads).
  lstm_scan<<<2, 800, SCAN_SMEM, stream>>>(Zf, Zb, whh1f, whh1b,
                                           h0 + 2 * Bb * Hh, c0 + 2 * Bb * Hh, out1, 0);

  // f head: [B*T,20] = out1 @ f_w^T + f_b (fp32 out).
  gemm_wmma<<<dim3(MTB / 16, 1), 32, 0, stream>>>(out1, LD1, fwh, LD1, f_b, out, FOUT, LD1 / 32, FOUT, 0);

  // g head: adjacent-pair concat then [B*(T-1),400] GEMM (fp32 out).
  build_gin<<<(MG * LDG + 255) / 256, 256, 0, stream>>>(out1, Gin);
  gemm_wmma<<<dim3(MG / 16, NP4H / 64), 32, 0, stream>>>(Gin, LDG, gwh, LDG, g_b,
                                                         out + (size_t)MTB * FOUT, GOUT,
                                                         LDG / 32, GOUT, 0);
}
